// QwenAttention_38139309589099
// MI455X (gfx1250) — compile-verified, matrix operand order checked by reading, not run on hardware
//
#include <hip/hip_runtime.h>

// ---- problem constants ----
#define HID_C 2048
#define NH_C 16
#define NKV_C 2
#define HD_C 128
#define GROUPS_C 8
#define B_C 2
#define S_C 2048
#define M_ROWS 4096           // B*S
#define LN_THETA 13.815510557964274f   // ln(1e6)
#define SCALE_C 0.08838834764831845f   // 1/sqrt(128)

typedef __bf16 v16bf __attribute__((ext_vector_type(16)));
typedef float  v8f   __attribute__((ext_vector_type(8)));

union Frag { v16bf v; unsigned int u[8]; uint4 q[2]; };

__device__ __forceinline__ unsigned short f2bf(float f) {
  unsigned int u = __float_as_uint(f);
  u += 0x7FFFu + ((u >> 16) & 1u);
  return (unsigned short)(u >> 16);
}

__device__ __forceinline__ v8f wmma_bf16(const Frag& a, const Frag& b, v8f c) {
  return __builtin_amdgcn_wmma_f32_16x16x32_bf16(false, a.v, false, b.v,
                                                 (short)0, c, false, false);
}

// Fragment = two 16B runs per lane: k = k0+half*8+{0..7} and k0+16+half*8+{0..7}
__device__ __forceinline__ Frag load_frag(const unsigned short* __restrict__ rowp,
                                          int k0, int half) {
  Frag f;
  f.q[0] = *(const uint4*)(rowp + k0 + half * 8);
  f.q[1] = *(const uint4*)(rowp + k0 + 16 + half * 8);
  return f;
}

// ---------------- elementwise converts ----------------
__global__ void cvt_bf16_kernel(const float* __restrict__ X,
                                unsigned short* __restrict__ Y, size_t n) {
  size_t i = (size_t)blockIdx.x * blockDim.x + threadIdx.x;
  if (i < n) Y[i] = f2bf(X[i]);
}

// RoPE on (B,S,nheads,HD) f32 -> bf16, pairs (i, i+64)
__global__ void rope_cvt_kernel(const float* __restrict__ X,
                                unsigned short* __restrict__ Y, int nheads) {
  size_t idx = (size_t)blockIdx.x * blockDim.x + threadIdx.x;
  size_t total = (size_t)B_C * S_C * nheads * (HD_C / 2);
  if (idx >= total) return;
  int i = (int)(idx % (HD_C / 2)); size_t t = idx / (HD_C / 2);
  int hh = (int)(t % nheads); t /= nheads;
  int s = (int)(t % S_C); size_t bb = t / S_C;
  float inv = __expf(-(2.0f * i / HD_C) * LN_THETA);
  float ang = (float)s * inv;
  float sn, cs; __sincosf(ang, &sn, &cs);
  size_t base = ((bb * S_C + s) * nheads + hh) * (size_t)HD_C;
  float x1 = X[base + i], x2 = X[base + 64 + i];
  Y[base + i]      = f2bf(x1 * cs - x2 * sn);
  Y[base + 64 + i] = f2bf(x2 * cs + x1 * sn);
}

// V (B,S,NKV,HD) f32 -> Vt (B,NKV,HD,S) bf16
__global__ void vtrans_kernel(const float* __restrict__ Vf,
                              unsigned short* __restrict__ Vt) {
  size_t idx = (size_t)blockIdx.x * blockDim.x + threadIdx.x;
  size_t total = (size_t)B_C * S_C * NKV_C * HD_C;
  if (idx >= total) return;
  int d = (int)(idx % HD_C); size_t t = idx / HD_C;
  int kv = (int)(t % NKV_C); t /= NKV_C;
  int s = (int)(t % S_C); size_t bb = t / S_C;
  Vt[(((bb * NKV_C + kv) * HD_C) + d) * (size_t)S_C + s] = f2bf(Vf[idx]);
}

// ---------------- WMMA bf16 GEMM: C = A(MxK) * W(NxK)^T + bias ----------------
// Wave computes 64x32 output (4 M-subtiles x 2 N-subtiles); block = 8 waves
// covering 256 (M) x 64 (N). No LDS, no barriers: fragments are direct b128s.
__global__ __launch_bounds__(256) void gemm_bf16_kernel(
    const unsigned short* __restrict__ A, const unsigned short* __restrict__ W,
    const float* __restrict__ bias, float* __restrict__ C, int M, int N, int K) {
  const int lane = threadIdx.x & 31;
  const int wave = threadIdx.x >> 5;
  const int half = lane >> 4;
  const int l16  = lane & 15;
  const int mbase = blockIdx.y * 256 + (wave >> 1) * 64;
  const int nbase = blockIdx.x * 64 + (wave & 1) * 32;

  const unsigned short* arow[4];
#pragma unroll
  for (int t = 0; t < 4; ++t)
    arow[t] = A + (size_t)(mbase + t * 16 + l16) * K;
  const unsigned short* wrow[2];
#pragma unroll
  for (int s2 = 0; s2 < 2; ++s2)
    wrow[s2] = W + (size_t)(nbase + s2 * 16 + l16) * K;

  v8f acc[4][2] = {};
  for (int k0 = 0; k0 < K; k0 += 32) {
    Frag fa[4], fb[2];
#pragma unroll
    for (int t = 0; t < 4; ++t) fa[t] = load_frag(arow[t], k0, half);
#pragma unroll
    for (int s2 = 0; s2 < 2; ++s2) fb[s2] = load_frag(wrow[s2], k0, half);
    if (k0 + 64 < K) {
      __builtin_prefetch(arow[0] + k0 + 64, 0, 1);
      __builtin_prefetch(wrow[0] + k0 + 64, 0, 1);
    }
#pragma unroll
    for (int t = 0; t < 4; ++t)
#pragma unroll
      for (int s2 = 0; s2 < 2; ++s2)
        acc[t][s2] = wmma_bf16(fa[t], fb[s2], acc[t][s2]);
  }

  float bv[2];
  bv[0] = bias ? bias[nbase + l16] : 0.0f;
  bv[1] = bias ? bias[nbase + 16 + l16] : 0.0f;
#pragma unroll
  for (int t = 0; t < 4; ++t)
#pragma unroll
    for (int s2 = 0; s2 < 2; ++s2)
#pragma unroll
      for (int r = 0; r < 8; ++r) {
        int row = mbase + t * 16 + r + half * 8;
        C[(size_t)row * N + nbase + s2 * 16 + l16] = acc[t][s2][r] + bv[s2];
      }
}

// ---------------- flash attention (one wave per 16-query tile) ----------------
__global__ __launch_bounds__(256) void attn_kernel(
    const unsigned short* __restrict__ Qb,  // (B,S,NH,HD)
    const unsigned short* __restrict__ Kb,  // (B,S,NKV,HD)
    const unsigned short* __restrict__ Vt,  // (B,NKV,HD,S)
    unsigned short* __restrict__ Ob) {      // (B,S,NH*HD)
  __shared__ unsigned int Pl32[8][16 * 16];  // per-wave 16x32 bf16 P tile
  const int lane = threadIdx.x & 31;
  const int wave = threadIdx.x >> 5;
  const int half = lane >> 4;
  const int l16  = lane & 15;
  const int b = blockIdx.z, h = blockIdx.y;
  const int qtile = blockIdx.x * 8 + wave;
  const int q0 = qtile * 16;
  const int kv = h / GROUPS_C;

  // Q fragments (4 K-chunks of 32 covering HD=128)
  Frag qf[4];
  const unsigned short* qrowp =
      Qb + (((size_t)b * S_C + q0 + l16) * NH_C + h) * HD_C;
#pragma unroll
  for (int c = 0; c < 4; ++c) qf[c] = load_frag(qrowp, c * 32, half);

  v8f o[8] = {};
  float m[8], lsum[8];
#pragma unroll
  for (int r = 0; r < 8; ++r) { m[r] = -1e30f; lsum[r] = 0.0f; }

  const int nkt = (q0 + 15) / 32 + 1;
  for (int kt = 0; kt < nkt; ++kt) {
    const int kbase = kt * 32;
    const unsigned short* k0p =
        Kb + (((size_t)b * S_C + kbase + l16) * NKV_C + kv) * HD_C;
    const unsigned short* k1p =
        Kb + (((size_t)b * S_C + kbase + 16 + l16) * NKV_C + kv) * HD_C;
    v8f s0 = {}, s1 = {};
#pragma unroll
    for (int c = 0; c < 4; ++c) {
      Frag kf0 = load_frag(k0p, c * 32, half);
      Frag kf1 = load_frag(k1p, c * 32, half);
      s0 = wmma_bf16(qf[c], kf0, s0);
      s1 = wmma_bf16(qf[c], kf1, s1);
    }
    // scale + causal mask + online softmax
    float corr[8];
    unsigned short* pw = (unsigned short*)&Pl32[wave][0];
#pragma unroll
    for (int r = 0; r < 8; ++r) {
      const int qr = q0 + r + half * 8;
      const int kc0 = kbase + l16, kc1 = kbase + 16 + l16;
      float v0 = s0[r] * SCALE_C + (kc0 <= qr ? 0.0f : -1e9f);
      float v1 = s1[r] * SCALE_C + (kc1 <= qr ? 0.0f : -1e9f);
      float vm = fmaxf(v0, v1);
      vm = fmaxf(vm, __shfl_xor(vm, 8, 32));
      vm = fmaxf(vm, __shfl_xor(vm, 4, 32));
      vm = fmaxf(vm, __shfl_xor(vm, 2, 32));
      vm = fmaxf(vm, __shfl_xor(vm, 1, 32));
      float nm = fmaxf(m[r], vm);
      corr[r] = __expf(m[r] - nm);
      m[r] = nm;
      float p0 = __expf(v0 - nm), p1 = __expf(v1 - nm);
      float ps = p0 + p1;
      ps += __shfl_xor(ps, 8, 32);
      ps += __shfl_xor(ps, 4, 32);
      ps += __shfl_xor(ps, 2, 32);
      ps += __shfl_xor(ps, 1, 32);
      lsum[r] = lsum[r] * corr[r] + ps;
      const int row = r + half * 8;
      pw[row * 32 + l16]      = f2bf(p0);
      pw[row * 32 + 16 + l16] = f2bf(p1);
    }
#pragma unroll
    for (int nt = 0; nt < 8; ++nt)
#pragma unroll
      for (int r = 0; r < 8; ++r) o[nt][r] *= corr[r];
    // re-layout P as A fragment via per-wave LDS (same-wave LDS is in-order)
    Frag pf;
    pf.q[0] = *(const uint4*)&Pl32[wave][l16 * 16 + half * 4];
    pf.q[1] = *(const uint4*)&Pl32[wave][l16 * 16 + 8 + half * 4];
    // P @ V
    const size_t vb = ((size_t)b * NKV_C + kv) * HD_C * (size_t)S_C;
#pragma unroll
    for (int nt = 0; nt < 8; ++nt) {
      const unsigned short* vcolp = Vt + vb + (size_t)(nt * 16 + l16) * S_C;
      Frag vf = load_frag(vcolp, kbase, half);
      o[nt] = wmma_bf16(pf, vf, o[nt]);
    }
  }
  // epilogue
  float inv[8];
#pragma unroll
  for (int r = 0; r < 8; ++r) inv[r] = 1.0f / lsum[r];
#pragma unroll
  for (int nt = 0; nt < 8; ++nt)
#pragma unroll
    for (int r = 0; r < 8; ++r) {
      int row = q0 + r + half * 8;
      Ob[((size_t)b * S_C + row) * (size_t)HID_C + h * HD_C + nt * 16 + l16] =
          f2bf(o[nt][r] * inv[r]);
    }
}

// ---------------- launcher ----------------
extern "C" void kernel_launch(void* const* d_in, const int* in_sizes, int n_in,
                              void* d_out, int out_size, void* d_ws, size_t ws_size,
                              hipStream_t stream) {
  (void)in_sizes; (void)n_in; (void)out_size; (void)ws_size;
  const float* hs  = (const float*)d_in[0];
  const float* q_w = (const float*)d_in[2];
  const float* q_b = (const float*)d_in[3];
  const float* k_w = (const float*)d_in[4];
  const float* k_b = (const float*)d_in[5];
  const float* v_w = (const float*)d_in[6];
  const float* v_b = (const float*)d_in[7];
  const float* o_w = (const float*)d_in[8];

  char* ws = (char*)d_ws;
  size_t off = 0;
  auto alloc = [&](size_t bytes) { size_t o = off; off += (bytes + 255) & ~(size_t)255; return o; };
  const size_t nHid = (size_t)B_C * S_C * HID_C;          // 8388608
  const size_t nKV  = (size_t)B_C * S_C * NKV_C * HD_C;   // 1048576
  unsigned short* Hb  = (unsigned short*)(ws + alloc(nHid * 2));
  unsigned short* Wqb = (unsigned short*)(ws + alloc((size_t)HID_C * HID_C * 2));
  unsigned short* Wkb = (unsigned short*)(ws + alloc((size_t)NKV_C * HD_C * HID_C * 2));
  unsigned short* Wvb = (unsigned short*)(ws + alloc((size_t)NKV_C * HD_C * HID_C * 2));
  unsigned short* Wob = (unsigned short*)(ws + alloc((size_t)HID_C * HID_C * 2));
  float* Qf = (float*)(ws + alloc(nHid * 4));
  float* Kf = (float*)(ws + alloc(nKV * 4));
  float* Vf = (float*)(ws + alloc(nKV * 4));
  unsigned short* Qb2 = (unsigned short*)(ws + alloc(nHid * 2));
  unsigned short* Kb2 = (unsigned short*)(ws + alloc(nKV * 2));
  unsigned short* Vt  = (unsigned short*)(ws + alloc(nKV * 2));
  unsigned short* At  = (unsigned short*)(ws + alloc(nHid * 2));

  const int TB = 256;
  cvt_bf16_kernel<<<(int)((nHid + TB - 1) / TB), TB, 0, stream>>>(hs, Hb, nHid);
  cvt_bf16_kernel<<<(int)(((size_t)HID_C * HID_C + TB - 1) / TB), TB, 0, stream>>>(q_w, Wqb, (size_t)HID_C * HID_C);
  cvt_bf16_kernel<<<(int)(((size_t)NKV_C * HD_C * HID_C + TB - 1) / TB), TB, 0, stream>>>(k_w, Wkb, (size_t)NKV_C * HD_C * HID_C);
  cvt_bf16_kernel<<<(int)(((size_t)NKV_C * HD_C * HID_C + TB - 1) / TB), TB, 0, stream>>>(v_w, Wvb, (size_t)NKV_C * HD_C * HID_C);
  cvt_bf16_kernel<<<(int)(((size_t)HID_C * HID_C + TB - 1) / TB), TB, 0, stream>>>(o_w, Wob, (size_t)HID_C * HID_C);

  // block tiles: 256 rows x 64 cols
  gemm_bf16_kernel<<<dim3(HID_C / 64, M_ROWS / 256), TB, 0, stream>>>(Hb, Wqb, q_b, Qf, M_ROWS, HID_C, HID_C);
  gemm_bf16_kernel<<<dim3((NKV_C * HD_C) / 64, M_ROWS / 256), TB, 0, stream>>>(Hb, Wkb, k_b, Kf, M_ROWS, NKV_C * HD_C, HID_C);
  gemm_bf16_kernel<<<dim3((NKV_C * HD_C) / 64, M_ROWS / 256), TB, 0, stream>>>(Hb, Wvb, v_b, Vf, M_ROWS, NKV_C * HD_C, HID_C);

  rope_cvt_kernel<<<(int)(((size_t)B_C * S_C * NH_C * 64 + TB - 1) / TB), TB, 0, stream>>>(Qf, Qb2, NH_C);
  rope_cvt_kernel<<<(int)(((size_t)B_C * S_C * NKV_C * 64 + TB - 1) / TB), TB, 0, stream>>>(Kf, Kb2, NKV_C);
  vtrans_kernel<<<(int)((nKV + TB - 1) / TB), TB, 0, stream>>>(Vf, Vt);

  attn_kernel<<<dim3(S_C / 128, NH_C, B_C), TB, 0, stream>>>(Qb2, Kb2, Vt, At);

  gemm_bf16_kernel<<<dim3(HID_C / 64, M_ROWS / 256), TB, 0, stream>>>(At, Wob, nullptr, (float*)d_out, M_ROWS, HID_C, HID_C);
}
